// MelSpectrogram_69526930588477
// MI455X (gfx1250) — matrix operand order repaired and checked by reading, not compiled, not run on hardware
//
#include <hip/hip_runtime.h>

// ---------------------------------------------------------------------------
// Mel-spectrogram for MI455X (gfx1250).
//   Stage 1: DFT GEMM via v_wmma_f32_16x16x32_bf16 with hi/lo split-bf16
//            (3 WMMAs per k-chunk per component ~= fp32 accuracy).
//   Stage 2: magnitude in registers -> LDS.
//   Stage 3: mel GEMM (80x513, tiny) + log1p in fp32 VALU.
// ---------------------------------------------------------------------------

#define FILTER_LEN 1024
#define HOP        256
#define N_MELS     80
#define CUTOFF     513           // FILTER_LEN/2 + 1
#define PADR       384           // (FILTER_LEN - HOP)/2
#define BATCH      8
#define TLEN       1048576
#define NFRAMES    4096          // frames per batch
#define FPB        16            // frames per workgroup
#define NTILES     33            // ceil(513/16)
#define KPAD       528           // NTILES*16 (padded bin rows in ws)
#define SEG_SAMP   (15*HOP + FILTER_LEN)   // 4864 samples staged per block
#define SEG_BYTES  10048         // 2*4864 + 16*(4864/256) = 10032, 16B-rounded
#define MAG_PITCH  17            // floats per bin row (conflict-free)
#define LDS_TOTAL  (2*SEG_BYTES + KPAD*MAG_PITCH*4)   // 56000 bytes

typedef __bf16 bf16x16 __attribute__((ext_vector_type(16)));
typedef float  f32x8   __attribute__((ext_vector_type(8)));

static __device__ __forceinline__ unsigned short f2bf(float f) {
    unsigned int u = __builtin_bit_cast(unsigned int, f);
    unsigned int r = u + 0x7fffu + ((u >> 16) & 1u);   // round-to-nearest-even
    return (unsigned short)(r >> 16);
}
static __device__ __forceinline__ float bf2f(unsigned short h) {
    unsigned int u = ((unsigned int)h) << 16;
    return __builtin_bit_cast(float, u);
}
static __device__ __forceinline__ bf16x16 mkfrag(uint4 a, uint4 b) {
    uint4 t[2] = {a, b};
    bf16x16 r;
    __builtin_memcpy(&r, t, 32);
    return r;
}

// LDS byte offset of sample j inside the staged segment (+16B pad / 256 samp
// so the 512B frame stride maps 16 lanes onto 16 distinct bank groups).
static __device__ __forceinline__ int seg_off(int j) {
    return 2 * j + 16 * (j >> 8);
}

// ---------------------------------------------------------------------------
// Kernel 1: fp32 bases -> bf16 hi/lo, row-padded to KPAD rows (zeros >= 513).
// ---------------------------------------------------------------------------
__global__ void mel_prep_bases(const float* __restrict__ br,
                               const float* __restrict__ bi,
                               unsigned short* __restrict__ rh,
                               unsigned short* __restrict__ rl,
                               unsigned short* __restrict__ ih,
                               unsigned short* __restrict__ il) {
    int idx = blockIdx.x * 256 + threadIdx.x;          // over KPAD*1024
    if (idx >= KPAD * FILTER_LEN) return;
    int bin = idx >> 10;
    int l   = idx & (FILTER_LEN - 1);
    float vr = (bin < CUTOFF) ? br[bin * FILTER_LEN + l] : 0.0f;
    float vi = (bin < CUTOFF) ? bi[bin * FILTER_LEN + l] : 0.0f;
    unsigned short h;
    h = f2bf(vr); rh[idx] = h; rl[idx] = f2bf(vr - bf2f(h));
    h = f2bf(vi); ih[idx] = h; il[idx] = f2bf(vi - bf2f(h));
}

// ---------------------------------------------------------------------------
// Kernel 2: fused DFT (WMMA) + magnitude + mel + log1p.
// grid = (frame_blocks=256, batch=8), block = 256 threads (8 waves).
// ---------------------------------------------------------------------------
__global__ void __launch_bounds__(256)
mel_spec_main(const float* __restrict__ audio,
              const float* __restrict__ mel,
              const unsigned short* __restrict__ rh,
              const unsigned short* __restrict__ rl,
              const unsigned short* __restrict__ ih,
              const unsigned short* __restrict__ il,
              float* __restrict__ out) {
    __shared__ __align__(16) unsigned char smem[LDS_TOTAL];
    unsigned char* seg_hi = smem;
    unsigned char* seg_lo = smem + SEG_BYTES;
    float* mag = (float*)(smem + 2 * SEG_BYTES);       // [KPAD][MAG_PITCH]

    const int tid  = threadIdx.x;
    const int lane = tid & 31;
    const int wave = tid >> 5;
    const int blk  = blockIdx.x;                       // frame block (0..255)
    const int b    = blockIdx.y;                       // batch

    // ---- stage reflected audio segment as bf16 hi/lo ----
    const int base = blk * (FPB * HOP) - PADR;
    const float* aud = audio + (size_t)b * TLEN;
    for (int j = tid; j < SEG_SAMP; j += 256) {
        int g = base + j;
        if (g < 0) g = -g;
        else if (g >= TLEN) g = 2 * TLEN - 2 - g;
        float x = aud[g];
        unsigned short h = f2bf(x);
        unsigned short l = f2bf(x - bf2f(h));
        int off = seg_off(j);
        *(unsigned short*)(seg_hi + off) = h;
        *(unsigned short*)(seg_lo + off) = l;
    }
    __syncthreads();

    // ---- DFT GEMM: per-wave 16(frames) x 16(bins) tiles over K=1024 ----
    const int m  = lane & 15;                          // frame row in tile
    const int kg = lane >> 4;                          // half-wave K group
    for (int nt = wave; nt < NTILES; nt += 8) {
        f32x8 accr = {0.f,0.f,0.f,0.f,0.f,0.f,0.f,0.f};
        f32x8 acci = {0.f,0.f,0.f,0.f,0.f,0.f,0.f,0.f};
        const int row = nt * 16 + m;                   // bin row (< KPAD)
        const unsigned short* prh = rh + (size_t)row * FILTER_LEN;
        const unsigned short* prl = rl + (size_t)row * FILTER_LEN;
        const unsigned short* pih = ih + (size_t)row * FILTER_LEN;
        const unsigned short* pil = il + (size_t)row * FILTER_LEN;

        for (int kb = 0; kb < FILTER_LEN / 32; ++kb) {
            const int k0 = kb * 32;
            // A fragment: elements 0..7 -> k = kg*8+e, 8..15 -> k = 16+kg*8+e
            const int s0 = m * HOP + k0 + kg * 8;
            const int s1 = s0 + 16;
            bf16x16 ahi = mkfrag(*(const uint4*)(seg_hi + seg_off(s0)),
                                 *(const uint4*)(seg_hi + seg_off(s1)));
            bf16x16 alo = mkfrag(*(const uint4*)(seg_lo + seg_off(s0)),
                                 *(const uint4*)(seg_lo + seg_off(s1)));
            // B fragments: lane holds column n=lane&15, elements 0..7 -> k = kg*8+e, 8..15 -> k = 16+kg*8+e
            const int bo = k0 + kg * 8;
            bf16x16 brh = mkfrag(*(const uint4*)(prh + bo),
                                 *(const uint4*)(prh + bo + 16));
            bf16x16 brl = mkfrag(*(const uint4*)(prl + bo),
                                 *(const uint4*)(prl + bo + 16));
            bf16x16 bih = mkfrag(*(const uint4*)(pih + bo),
                                 *(const uint4*)(pih + bo + 16));
            bf16x16 bil = mkfrag(*(const uint4*)(pil + bo),
                                 *(const uint4*)(pil + bo + 16));
            // split-bf16: hi*hi + hi*lo + lo*hi  (real and imag)
            accr = __builtin_amdgcn_wmma_f32_16x16x32_bf16(false, ahi, false, brh, (short)0, accr, false, false);
            accr = __builtin_amdgcn_wmma_f32_16x16x32_bf16(false, ahi, false, brl, (short)0, accr, false, false);
            accr = __builtin_amdgcn_wmma_f32_16x16x32_bf16(false, alo, false, brh, (short)0, accr, false, false);
            acci = __builtin_amdgcn_wmma_f32_16x16x32_bf16(false, ahi, false, bih, (short)0, acci, false, false);
            acci = __builtin_amdgcn_wmma_f32_16x16x32_bf16(false, ahi, false, bil, (short)0, acci, false, false);
            acci = __builtin_amdgcn_wmma_f32_16x16x32_bf16(false, alo, false, bih, (short)0, acci, false, false);
        }
        // D layout: element v -> (frame = v + 8*kg, bin col = lane&15)
        #pragma unroll
        for (int v = 0; v < 8; ++v) {
            int fr = v + 8 * kg;
            float r = accr[v], i = acci[v];
            mag[row * MAG_PITCH + fr] = sqrtf(r * r + i * i);
        }
    }
    __syncthreads();

    // ---- mel projection + log1p (fp32 VALU; ~1% of FLOPs) ----
    const int f  = tid & 15;                           // local frame
    const int mb = tid >> 4;                           // mel row base (0..15)
    float acc[5] = {0.f, 0.f, 0.f, 0.f, 0.f};
    for (int k = 0; k < CUTOFF; ++k) {
        float mv = mag[k * MAG_PITCH + f];
        #pragma unroll
        for (int r = 0; r < 5; ++r)
            acc[r] += mel[(mb + 16 * r) * CUTOFF + k] * mv;
    }
    const int fg = blk * FPB + f;
    #pragma unroll
    for (int r = 0; r < 5; ++r) {
        int mrow = mb + 16 * r;
        out[(size_t)b * N_MELS * NFRAMES + (size_t)mrow * NFRAMES + fg] =
            log1pf(10000.0f * acc[r]);
    }
}

// ---------------------------------------------------------------------------
extern "C" void kernel_launch(void* const* d_in, const int* in_sizes, int n_in,
                              void* d_out, int out_size, void* d_ws, size_t ws_size,
                              hipStream_t stream) {
    const float* audio   = (const float*)d_in[0];   // [8, 1048576]
    const float* basis_r = (const float*)d_in[1];   // [513, 1024]
    const float* basis_i = (const float*)d_in[2];   // [513, 1024]
    const float* mel     = (const float*)d_in[3];   // [80, 513]
    float* out = (float*)d_out;                     // [8, 80, 4096]

    const size_t arr = (size_t)KPAD * FILTER_LEN;   // elems per bf16 array
    unsigned short* rh = (unsigned short*)d_ws;
    unsigned short* rl = rh + arr;
    unsigned short* ih = rl + arr;
    unsigned short* il = ih + arr;                  // ws usage: ~4.3 MB

    mel_prep_bases<<<(int)((KPAD * FILTER_LEN + 255) / 256), 256, 0, stream>>>(
        basis_r, basis_i, rh, rl, ih, il);

    dim3 grid(NFRAMES / FPB, BATCH);                // (256, 8)
    mel_spec_main<<<grid, 256, 0, stream>>>(audio, mel, rh, rl, ih, il, out);
}
